// MultiHeadAttention_61383672594590
// MI455X (gfx1250) — matrix operand order repaired and checked by reading, compiled, not run on hardware
//
#include <hip/hip_runtime.h>
#include <hip/hip_bf16.h>

typedef _Float16 half_t;
typedef __attribute__((ext_vector_type(16))) _Float16 v16h;
typedef __attribute__((ext_vector_type(8)))  _Float16 v8h;
typedef __attribute__((ext_vector_type(8)))  float    v8f;

#define SEQ   2048
#define DMODEL 1024
#define NHEAD 16
#define HDIM  64
#define BATCH 4

static __device__ __forceinline__ v16h combine16(v8h lo, v8h hi) {
    v16h r;
#pragma unroll
    for (int i = 0; i < 8; ++i) { r[i] = lo[i]; r[i + 8] = hi[i]; }
    return r;
}

static __device__ __forceinline__ float half_reduce_max16(float v) {
#pragma unroll
    for (int off = 1; off < 16; off <<= 1) v = fmaxf(v, __shfl_xor(v, off, 32));
    return v;
}
static __device__ __forceinline__ float half_reduce_sum16(float v) {
#pragma unroll
    for (int off = 1; off < 16; off <<= 1) v += __shfl_xor(v, off, 32);
    return v;
}

// ---------------- conversion kernels ----------------
__global__ __launch_bounds__(256) void f32_to_f16_kernel(const float* __restrict__ in,
                                                         half_t* __restrict__ out, size_t n) {
    size_t i = (size_t)blockIdx.x * blockDim.x + threadIdx.x;
    size_t stride = (size_t)gridDim.x * blockDim.x;
    for (; i < n; i += stride) out[i] = (half_t)in[i];
}

// in: [R][C] f32 row-major  ->  out: [C][R] f16 row-major (i.e. transposed)
__global__ __launch_bounds__(256) void transpose_f32_to_f16_kernel(const float* __restrict__ in,
                                                                   half_t* __restrict__ out,
                                                                   int R, int C) {
    size_t i = (size_t)blockIdx.x * blockDim.x + threadIdx.x;
    size_t n = (size_t)R * C;
    size_t stride = (size_t)gridDim.x * blockDim.x;
    for (; i < n; i += stride) {
        int r = (int)(i / C);
        int c = (int)(i % C);
        out[(size_t)c * R + r] = (half_t)in[i];
    }
}

// ---------------- WMMA GEMM ----------------
// C[M,N] = A[M,K] (f16, row-major) * BT[N,K] (f16, B transposed) + bias
// mode 0: store f32 to Cout
// mode 1: QKV epilogue: scatter to Qb/Kb ([B,H,S,HD] f16, Q scaled 1/8) and Vt ([B,H,HD,S] f16)
// Block: 256 threads (8 waves), block tile 128(M) x 128(N), wave tile 64 x 32, K step 32.
__global__ __launch_bounds__(256) void gemm_wmma_kernel(const half_t* __restrict__ A,
                                                        const half_t* __restrict__ BT,
                                                        const float* __restrict__ bias,
                                                        float* __restrict__ Cout,
                                                        half_t* __restrict__ Qb,
                                                        half_t* __restrict__ Kb,
                                                        half_t* __restrict__ Vt,
                                                        int M, int N, int K, int mode) {
    const int lane = threadIdx.x & 31;
    const int wave = threadIdx.x >> 5;
    const int wm = wave & 1;       // 0..1
    const int wn = wave >> 1;      // 0..3
    const int m_base = blockIdx.y * 128 + wm * 64;
    const int n_base = blockIdx.x * 128 + wn * 32;
    const int ml = lane & 15;      // M (or N) within 16-tile
    const int hi = lane >> 4;      // lane-half

    v8f acc[4][2] = {};

    for (int k0 = 0; k0 < K; k0 += 32) {
        v16h af[4], bf[2];
#pragma unroll
        for (int mt = 0; mt < 4; ++mt) {
            const half_t* pa = A + (size_t)(m_base + mt * 16 + ml) * K + k0;
            v8h lo = *(const v8h*)(pa + 8 * hi);
            v8h hb = *(const v8h*)(pa + 16 + 8 * hi);
            af[mt] = combine16(lo, hb);
        }
#pragma unroll
        for (int nt = 0; nt < 2; ++nt) {
            const half_t* pb = BT + (size_t)(n_base + nt * 16 + ml) * K + k0;
            v8h lo = *(const v8h*)(pb + 8 * hi);
            v8h hb = *(const v8h*)(pb + 16 + 8 * hi);
            bf[nt] = combine16(lo, hb);
        }
#pragma unroll
        for (int mt = 0; mt < 4; ++mt)
#pragma unroll
            for (int nt = 0; nt < 2; ++nt)
                acc[mt][nt] = __builtin_amdgcn_wmma_f32_16x16x32_f16(
                    false, af[mt], false, bf[nt], (short)0, acc[mt][nt], false, false);
    }

#pragma unroll
    for (int mt = 0; mt < 4; ++mt) {
#pragma unroll
        for (int nt = 0; nt < 2; ++nt) {
            const int col = n_base + nt * 16 + ml;
            const float bv = bias[col];
#pragma unroll
            for (int r = 0; r < 8; ++r) {
                const int row = m_base + mt * 16 + r + 8 * hi;
                float v = acc[mt][nt][r] + bv;
                if (mode == 0) {
                    Cout[(size_t)row * N + col] = v;
                } else {
                    const int which = col >> 10;          // 0=q 1=k 2=v
                    const int d = col & (DMODEL - 1);
                    const int h = d >> 6;
                    const int hd = d & (HDIM - 1);
                    const int b = row >> 11;              // /SEQ
                    const int s = row & (SEQ - 1);
                    if (which == 0) {
                        Qb[(((size_t)(b * NHEAD + h)) * SEQ + s) * HDIM + hd] =
                            (half_t)(v * 0.125f);         // 1/sqrt(64) folded in
                    } else if (which == 1) {
                        Kb[(((size_t)(b * NHEAD + h)) * SEQ + s) * HDIM + hd] = (half_t)v;
                    } else {
                        Vt[(((size_t)(b * NHEAD + h)) * HDIM + hd) * SEQ + s] = (half_t)v;
                    }
                }
            }
        }
    }
}

// ---------------- causal flash attention ----------------
// grid: (S/64, B*H); block 128 (4 waves); each wave owns a 16-query tile.
// Qb/Kb: [B,H,S,HD] f16 (Q pre-scaled), Vt: [B,H,HD,S] f16, out attn_h: [B,S,D] f16.
__global__ __launch_bounds__(128) void flash_attn_kernel(const half_t* __restrict__ Qb,
                                                         const half_t* __restrict__ Kb,
                                                         const half_t* __restrict__ Vt,
                                                         half_t* __restrict__ attn_h) {
    __shared__ half_t lds[4 * 16 * 32];   // per-wave 16x32 P staging tile
    const int lane = threadIdx.x & 31;
    const int wave = threadIdx.x >> 5;
    const int ml = lane & 15;
    const int hi = lane >> 4;
    const int bh = blockIdx.y;                  // b*NHEAD + h
    const int q0 = (blockIdx.x * 4 + wave) * 16;

    // Q A-fragments (16 x 64 => two K=32 chunks)
    const half_t* Qp = Qb + ((size_t)bh * SEQ + q0) * HDIM;
    v16h qf[2];
#pragma unroll
    for (int c = 0; c < 2; ++c) {
        const half_t* pq = Qp + (size_t)ml * HDIM + 32 * c;
        v8h lo = *(const v8h*)(pq + 8 * hi);
        v8h hb = *(const v8h*)(pq + 16 + 8 * hi);
        qf[c] = combine16(lo, hb);
    }

    v8f acc[4] = {};
    float mi[8], li[8];
#pragma unroll
    for (int r = 0; r < 8; ++r) { mi[r] = -__builtin_inff(); li[r] = 0.0f; }

    half_t* pbase = lds + wave * (16 * 32);
    const int nkb = (q0 + 15) / 32 + 1;         // causal: key blocks of 32

    for (int kb = 0; kb < nkb; ++kb) {
        const int key0 = kb * 32;

        // S = Q @ K^T : two 16x16 score tiles (keys key0..+15, +16..+31)
        v8f sc[2] = {};
#pragma unroll
        for (int nt = 0; nt < 2; ++nt) {
#pragma unroll
            for (int c = 0; c < 2; ++c) {
                const half_t* pk = Kb + ((size_t)bh * SEQ + key0 + nt * 16 + ml) * HDIM
                                   + 32 * c + 8 * hi;
                v8h lo = *(const v8h*)(pk);
                v8h hb = *(const v8h*)(pk + 16);
                v16h kf = combine16(lo, hb);
                sc[nt] = __builtin_amdgcn_wmma_f32_16x16x32_f16(
                    false, qf[c], false, kf, (short)0, sc[nt], false, false);
            }
        }

        // causal mask on diagonal blocks
        if (key0 + 31 > q0) {
#pragma unroll
            for (int nt = 0; nt < 2; ++nt) {
                const int key = key0 + nt * 16 + ml;
#pragma unroll
                for (int r = 0; r < 8; ++r) {
                    const int qq = q0 + r + 8 * hi;
                    if (key > qq) sc[nt][r] = -__builtin_inff();
                }
            }
        }

        // online softmax
        float alpha[8];
#pragma unroll
        for (int r = 0; r < 8; ++r) {
            float rm = half_reduce_max16(fmaxf(sc[0][r], sc[1][r]));
            float mn = fmaxf(mi[r], rm);
            alpha[r] = __expf(mi[r] - mn);      // exp(-inf)=0 on first block
            float p0 = __expf(sc[0][r] - mn);
            float p1 = __expf(sc[1][r] - mn);
            sc[0][r] = p0;
            sc[1][r] = p1;
            float rs = half_reduce_sum16(p0 + p1);
            li[r] = li[r] * alpha[r] + rs;
            mi[r] = mn;
        }
#pragma unroll
        for (int nt = 0; nt < 4; ++nt)
#pragma unroll
            for (int r = 0; r < 8; ++r) acc[nt][r] *= alpha[r];

        // P (16x32) -> LDS -> A-fragment relayout (wave-private, no barrier needed)
#pragma unroll
        for (int nt = 0; nt < 2; ++nt)
#pragma unroll
            for (int r = 0; r < 8; ++r)
                pbase[(r + 8 * hi) * 32 + nt * 16 + ml] = (half_t)sc[nt][r];
        asm volatile("s_wait_dscnt 0" ::: "memory");
        const half_t* pr = pbase + ml * 32;
        v8h plo = *(const v8h*)(pr + 8 * hi);
        v8h phb = *(const v8h*)(pr + 16 + 8 * hi);
        v16h pf = combine16(plo, phb);

        // acc += P @ V  (V transposed layout => contiguous B fragments)
#pragma unroll
        for (int nt = 0; nt < 4; ++nt) {
            const half_t* pv = Vt + ((size_t)bh * HDIM + nt * 16 + ml) * SEQ + key0 + 8 * hi;
            v8h lo = *(const v8h*)(pv);
            v8h hb = *(const v8h*)(pv + 16);
            v16h vf = combine16(lo, hb);
            acc[nt] = __builtin_amdgcn_wmma_f32_16x16x32_f16(
                false, pf, false, vf, (short)0, acc[nt], false, false);
        }
    }

    // normalize and store to attn_h [B,S,D] f16
    const int b = bh >> 4;
    const int h = bh & (NHEAD - 1);
#pragma unroll
    for (int nt = 0; nt < 4; ++nt) {
#pragma unroll
        for (int r = 0; r < 8; ++r) {
            const int tok = q0 + r + 8 * hi;
            const int col = h * HDIM + nt * 16 + ml;
            attn_h[((size_t)b * SEQ + tok) * DMODEL + col] = (half_t)(acc[nt][r] / li[r]);
        }
    }
}

// ---------------- launcher ----------------
extern "C" void kernel_launch(void* const* d_in, const int* in_sizes, int n_in,
                              void* d_out, int out_size, void* d_ws, size_t ws_size,
                              hipStream_t stream) {
    const float* x     = (const float*)d_in[0];   // [B,S,D]
    const float* w_qkv = (const float*)d_in[1];   // [D,3D]
    const float* b_qkv = (const float*)d_in[2];   // [3D]
    const float* w_out = (const float*)d_in[3];   // [D,D]
    const float* b_out = (const float*)d_in[4];   // [D]
    float* out = (float*)d_out;

    const size_t nX   = (size_t)BATCH * SEQ * DMODEL;            // 8.4M
    const size_t nWq  = (size_t)DMODEL * 3 * DMODEL;             // 3.1M
    const size_t nWo  = (size_t)DMODEL * DMODEL;                 // 1.0M
    const size_t nHd  = (size_t)BATCH * NHEAD * SEQ * HDIM;      // 8.4M

    char* ws = (char*)d_ws;
    size_t off = 0;
    half_t* xh     = (half_t*)(ws + off); off += nX  * sizeof(half_t);
    half_t* wqkvT  = (half_t*)(ws + off); off += nWq * sizeof(half_t);
    half_t* woutT  = (half_t*)(ws + off); off += nWo * sizeof(half_t);
    half_t* Qb     = (half_t*)(ws + off); off += nHd * sizeof(half_t);
    half_t* Kb     = (half_t*)(ws + off); off += nHd * sizeof(half_t);
    half_t* Vt     = (half_t*)(ws + off); off += nHd * sizeof(half_t);
    half_t* attn_h = (half_t*)(ws + off); off += nX  * sizeof(half_t);

    // 1) precision conversion / weight transposition
    f32_to_f16_kernel<<<4096, 256, 0, stream>>>(x, xh, nX);
    transpose_f32_to_f16_kernel<<<4096, 256, 0, stream>>>(w_qkv, wqkvT, DMODEL, 3 * DMODEL);
    transpose_f32_to_f16_kernel<<<2048, 256, 0, stream>>>(w_out, woutT, DMODEL, DMODEL);

    // 2) QKV projection with fused head-split / scale / V-transpose epilogue
    {
        dim3 grid((3 * DMODEL) / 128, (BATCH * SEQ) / 128);   // (24, 64)
        gemm_wmma_kernel<<<grid, 256, 0, stream>>>(xh, wqkvT, b_qkv, nullptr,
                                                   Qb, Kb, Vt,
                                                   BATCH * SEQ, 3 * DMODEL, DMODEL, 1);
    }

    // 3) causal flash attention
    {
        dim3 grid(SEQ / 64, BATCH * NHEAD);                   // (32, 64)
        flash_attn_kernel<<<grid, 128, 0, stream>>>(Qb, Kb, Vt, attn_h);
    }

    // 4) output projection
    {
        dim3 grid(DMODEL / 128, (BATCH * SEQ) / 128);         // (8, 64)
        gemm_wmma_kernel<<<grid, 256, 0, stream>>>(attn_h, woutT, b_out, out,
                                                   nullptr, nullptr, nullptr,
                                                   BATCH * SEQ, DMODEL, DMODEL, 0);
    }
}
